// EPA2D_84482006713040
// MI455X (gfx1250) — matrix operand order repaired and checked by reading, not compiled, not run on hardware
//
#include <hip/hip_runtime.h>
#include <math.h>

// ---------------------------------------------------------------------------
// EPA2D fused pipeline for gfx1250 (MI455X). bf16 WMMA with f32 accumulation.
// Round 3: explicit two-stage ping-pong (no register-copy rotation) in the
// K=1024 GEMM loops; packed 16B transposed stores in QKVV.
// ---------------------------------------------------------------------------

typedef __bf16 bf16;
typedef __attribute__((ext_vector_type(16))) __bf16 bf16x16;
typedef __attribute__((ext_vector_type(8)))  __bf16 bf16x8;
typedef __attribute__((ext_vector_type(8)))  float   f32x8;

__device__ __forceinline__ int lane_id() { return (int)(threadIdx.x & 31u); }

// A-fragment (16x32 bf16, M x K), row-major source, row stride ld (elements).
// Lane l holds row l&15; K chunks [8h,8h+8) and [16+8h,16+8h+8), h = l>>4.
__device__ __forceinline__ bf16x16 load_a(const bf16* p, int ld) {
  int l = lane_id(); int row = l & 15; int half = l >> 4;
  const bf16* r0 = p + (size_t)row * ld + half * 8;
  bf16x8 lo = *(const bf16x8*)(r0);
  bf16x8 hi = *(const bf16x8*)(r0 + 16);
  return __builtin_shufflevector(lo, hi, 0,1,2,3,4,5,6,7,8,9,10,11,12,13,14,15);
}

// A-fragment built from an f32 LDS tile (convert on the fly).
__device__ __forceinline__ bf16x16 load_a_f32(const float* p, int ld) {
  int l = lane_id(); int row = l & 15; int half = l >> 4;
  const float* r0 = p + (size_t)row * ld + half * 8;
  bf16x16 out;
#pragma unroll
  for (int i = 0; i < 8; i++) { out[i] = (bf16)r0[i]; out[8 + i] = (bf16)r0[16 + i]; }
  return out;
}

// B-fragment (32x16 bf16, K x N) supplied as B^T row-major (weight rows):
// lane l holds column l&15; K chunks [8h,8h+8) and [16+8h,16+8h+8), h = l>>4.
__device__ __forceinline__ bf16x16 load_b(const bf16* bt, int ld) {
  int l = lane_id(); int col = l & 15; int half = l >> 4;
  const bf16* r0 = bt + (size_t)col * ld + half * 8;
  bf16x8 lo = *(const bf16x8*)(r0);
  bf16x8 hi = *(const bf16x8*)(r0 + 16);
  return __builtin_shufflevector(lo, hi, 0,1,2,3,4,5,6,7,8,9,10,11,12,13,14,15);
}

__device__ __forceinline__ f32x8 wmma_bf16(bf16x16 a, bf16x16 b, f32x8 c) {
  return __builtin_amdgcn_wmma_f32_16x16x32_bf16(false, a, false, b, (short)0, c,
                                                 false, false);
}

// ---------------------------------------------------------------------------
// f32 -> bf16 conversion (grid-stride)
// ---------------------------------------------------------------------------
__global__ void cvt_f32_bf16(const float* __restrict__ in, bf16* __restrict__ out,
                             int n) {
  int i = blockIdx.x * blockDim.x + threadIdx.x;
  int stride = gridDim.x * blockDim.x;
  for (; i < n; i += stride) out[i] = (bf16)in[i];
}

// ---------------------------------------------------------------------------
// QKVV GEMM: [8192,1024] x [4096,1024]^T, scattered into q/qT/k/kT/vca/vsa.
// Per-wave 32x64 tile (8 accumulators; one 64-wide strip == one head).
// Two-stage ping-pong over K: compute E while O's loads are in flight.
// ---------------------------------------------------------------------------
__global__ __launch_bounds__(256) void qkvv_kernel(
    const bf16* __restrict__ xb,   // [8192,1024]
    const bf16* __restrict__ wb,   // [4096,1024]
    bf16* __restrict__ q,  bf16* __restrict__ qT,
    bf16* __restrict__ k,  bf16* __restrict__ kT,
    bf16* __restrict__ vca, bf16* __restrict__ vsa) {
  const int K = 1024, N = 1024, D = 64, Hh = 16;
  int wave = threadIdx.x >> 5;
  int job = blockIdx.x * 8 + wave;        // 16384 jobs
  int mt = job >> 6;                      // 256 M-tiles of 32 rows
  int n0 = (job & 63) * 64;               // output strip (one head)
  const bf16* A0 = xb + (size_t)(mt * 32) * K;
  const bf16* A1 = A0 + (size_t)16 * K;
  const bf16* Bp = wb + (size_t)n0 * K;
  f32x8 acc[8] = {};
  bf16x16 aE0, aE1, bE[4], aO0, aO1, bO[4];
  aE0 = load_a(A0, K);
  aE1 = load_a(A1, K);
#pragma unroll
  for (int t = 0; t < 4; t++) bE[t] = load_b(Bp + (size_t)(t * 16) * K, K);
  aO0 = load_a(A0 + 32, K);
  aO1 = load_a(A1 + 32, K);
#pragma unroll
  for (int t = 0; t < 4; t++) bO[t] = load_b(Bp + (size_t)(t * 16) * K + 32, K);
  for (int kk = 0; kk < K - 64; kk += 64) {
    __builtin_prefetch((const void*)(A0 + kk + 128), 0, 1);
    __builtin_prefetch((const void*)(A1 + kk + 128), 0, 1);
#pragma unroll
    for (int t = 0; t < 4; t++) acc[t]     = wmma_bf16(aE0, bE[t], acc[t]);
#pragma unroll
    for (int t = 0; t < 4; t++) acc[4 + t] = wmma_bf16(aE1, bE[t], acc[4 + t]);
    aE0 = load_a(A0 + kk + 64, K);
    aE1 = load_a(A1 + kk + 64, K);
#pragma unroll
    for (int t = 0; t < 4; t++) bE[t] = load_b(Bp + (size_t)(t * 16) * K + kk + 64, K);
#pragma unroll
    for (int t = 0; t < 4; t++) acc[t]     = wmma_bf16(aO0, bO[t], acc[t]);
#pragma unroll
    for (int t = 0; t < 4; t++) acc[4 + t] = wmma_bf16(aO1, bO[t], acc[4 + t]);
    aO0 = load_a(A0 + kk + 96, K);
    aO1 = load_a(A1 + kk + 96, K);
#pragma unroll
    for (int t = 0; t < 4; t++) bO[t] = load_b(Bp + (size_t)(t * 16) * K + kk + 96, K);
  }
#pragma unroll
  for (int t = 0; t < 4; t++) acc[t]     = wmma_bf16(aE0, bE[t], acc[t]);
#pragma unroll
  for (int t = 0; t < 4; t++) acc[4 + t] = wmma_bf16(aE1, bE[t], acc[4 + t]);
#pragma unroll
  for (int t = 0; t < 4; t++) acc[t]     = wmma_bf16(aO0, bO[t], acc[t]);
#pragma unroll
  for (int t = 0; t < 4; t++) acc[4 + t] = wmma_bf16(aO1, bO[t], acc[4 + t]);

  int l = lane_id(); int half = l >> 4; int cl = l & 15;
  int which = n0 >> 10;                   // 0:q 1:k 2:v_ca 3:v_sa
  int h = (n0 & 1023) >> 6;
  int bidx = (mt * 32) >> 10;             // batch (constant over the tile)
  // scalar stores in the [.,N,d] layout
  bf16* dstN = (which == 0) ? q : (which == 1) ? k : (which == 2) ? vca : vsa;
#pragma unroll
  for (int sub = 0; sub < 2; sub++) {
#pragma unroll
    for (int t = 0; t < 4; t++) {
      int dd = t * 16 + cl;
#pragma unroll
      for (int r = 0; r < 8; r++) {
        int m = mt * 32 + sub * 16 + r + 8 * half;
        int n = m & 1023;
        dstN[(((size_t)(bidx * Hh + h)) * N + n) * D + dd] = (bf16)acc[sub * 4 + t][r];
      }
    }
  }
  // packed 16B stores in the transposed [.,d,N] layout (q and k only):
  // for fixed (sub,t,half) the 8 accumulator rows are 8 consecutive n.
  if (which < 2) {
    bf16* dstT = (which == 0) ? qT : kT;
#pragma unroll
    for (int sub = 0; sub < 2; sub++) {
#pragma unroll
      for (int t = 0; t < 4; t++) {
        int dd = t * 16 + cl;
        bf16x8 pk;
#pragma unroll
        for (int r = 0; r < 8; r++) pk[r] = (bf16)acc[sub * 4 + t][r];
        int nrow = (mt * 32 + sub * 16 + 8 * half) & 1023;
        *(bf16x8*)(dstT + (((size_t)(bidx * Hh + h)) * D + dd) * (size_t)N + nrow) = pk;
      }
    }
  }
}

// ---------------------------------------------------------------------------
// Channel (transposed) attention: per (b,h): S = q^T k [64x64, K=1024],
// softmax rows, x_c = P @ v_ca^T [64x1024]; store in torch-faithful
// [B, d, H, N] layout so the output GEMM reads contiguous rows.
// ---------------------------------------------------------------------------
__global__ __launch_bounds__(256) void chan_attn_kernel(
    const bf16* __restrict__ qT,   // [BH,64,1024]
    const bf16* __restrict__ kT,   // [BH,64,1024]
    const bf16* __restrict__ vca,  // [BH,1024,64]
    const float* __restrict__ temp1,
    bf16* __restrict__ xc) {       // flat [B][64][16][1024]
  __shared__ float S[64][64];
  __shared__ bf16  P[64][64];
  int bh = blockIdx.x; int b = bh >> 4, h = bh & 15;
  float t1 = temp1[h];
  int wave = threadIdx.x >> 5; int l = lane_id(); int half = l >> 4; int cl = l & 15;
  const bf16* qb = qT + (size_t)bh * 64 * 1024;
  const bf16* kb = kT + (size_t)bh * 64 * 1024;
  // phase 1: scores. Wave w owns tiles (ti, tj0) and (ti, tj0+1) sharing the
  // A fragment; two-stage ping-pong over K.
  {
    int ti = wave >> 1, tj0 = (wave & 1) * 2;
    const bf16* arow  = qb + (size_t)(ti * 16) * 1024;
    const bf16* brow0 = kb + (size_t)(tj0 * 16) * 1024;
    const bf16* brow1 = brow0 + (size_t)16 * 1024;
    f32x8 acc0 = {}, acc1 = {};
    bf16x16 aE  = load_a(arow, 1024);
    bf16x16 bE0 = load_b(brow0, 1024);
    bf16x16 bE1 = load_b(brow1, 1024);
    bf16x16 aO  = load_a(arow + 32, 1024);
    bf16x16 bO0 = load_b(brow0 + 32, 1024);
    bf16x16 bO1 = load_b(brow1 + 32, 1024);
    for (int kk = 0; kk < 1024 - 64; kk += 64) {
      __builtin_prefetch((const void*)(arow + kk + 128), 0, 1);
      acc0 = wmma_bf16(aE, bE0, acc0);
      acc1 = wmma_bf16(aE, bE1, acc1);
      aE  = load_a(arow + kk + 64, 1024);
      bE0 = load_b(brow0 + kk + 64, 1024);
      bE1 = load_b(brow1 + kk + 64, 1024);
      acc0 = wmma_bf16(aO, bO0, acc0);
      acc1 = wmma_bf16(aO, bO1, acc1);
      aO  = load_a(arow + kk + 96, 1024);
      bO0 = load_b(brow0 + kk + 96, 1024);
      bO1 = load_b(brow1 + kk + 96, 1024);
    }
    acc0 = wmma_bf16(aE, bE0, acc0);
    acc1 = wmma_bf16(aE, bE1, acc1);
    acc0 = wmma_bf16(aO, bO0, acc0);
    acc1 = wmma_bf16(aO, bO1, acc1);
#pragma unroll
    for (int r = 0; r < 8; r++) {
      S[ti * 16 + r + 8 * half][tj0 * 16 + cl]       = acc0[r] * t1;
      S[ti * 16 + r + 8 * half][(tj0 + 1) * 16 + cl] = acc1[r] * t1;
    }
  }
  __syncthreads();
  // phase 2: softmax (64 rows, one per thread)
  if (threadIdx.x < 64) {
    int row = threadIdx.x;
    float m = -INFINITY;
    for (int j = 0; j < 64; j++) m = fmaxf(m, S[row][j]);
    float s = 0.f;
    for (int j = 0; j < 64; j++) { float e = __expf(S[row][j] - m); S[row][j] = e; s += e; }
    float inv = 1.f / s;
    for (int j = 0; j < 64; j++) P[row][j] = (bf16)(S[row][j] * inv);
  }
  __syncthreads();
  // phase 3: x_c = P @ v_ca^T  (256 tiles, 32 per wave)
  const bf16* vb = vca + (size_t)bh * 1024 * 64;
  for (int tile = wave; tile < 256; tile += 8) {
    int ti = tile >> 6, tn = tile & 63;
    f32x8 acc = {};
#pragma unroll
    for (int kk = 0; kk < 64; kk += 32) {
      bf16x16 a = load_a(&P[ti * 16][0] + kk, 64);
      bf16x16 bb = load_b(vb + (size_t)(tn * 16) * 64 + kk, 64);
      acc = wmma_bf16(a, bb, acc);
    }
#pragma unroll
    for (int r = 0; r < 8; r++) {
      int i = ti * 16 + r + 8 * half;
      int n = tn * 16 + cl;
      xc[(((size_t)b * 64 + i) * 16 + h) * 1024 + n] = (bf16)acc[r];
    }
  }
}

// ---------------------------------------------------------------------------
// E/F projections: dst = src @ W^T + bias, per (b,h) [1024x64, K=64].
// transposeOut=1 writes [d,N] (for v_s, consumed as B^T by the PV GEMM).
// ---------------------------------------------------------------------------
__global__ __launch_bounds__(256) void proj_ef_kernel(
    const bf16* __restrict__ src,   // [BH,1024,64]
    const bf16* __restrict__ w,     // [64,64] row-major (== B^T rows)
    const float* __restrict__ bias, // [64]
    bf16* __restrict__ dst, int transposeOut) {
  int wave = threadIdx.x >> 5;
  int job = blockIdx.x * 8 + wave;  // 8192 = 128 bh * 64 M-tiles
  int bh = job >> 6; int mt = job & 63;
  const bf16* A = src + ((size_t)bh * 1024 + mt * 16) * 64;
  f32x8 acc[4] = {};
#pragma unroll
  for (int kk = 0; kk < 64; kk += 32) {
    bf16x16 a = load_a(A + kk, 64);
#pragma unroll
    for (int t = 0; t < 4; t++) {
      bf16x16 bb = load_b(w + (size_t)(t * 16) * 64 + kk, 64);
      acc[t] = wmma_bf16(a, bb, acc[t]);
    }
  }
  int l = lane_id(); int half = l >> 4; int cl = l & 15;
#pragma unroll
  for (int t = 0; t < 4; t++) {
    int dd = t * 16 + cl;
    float bv = bias[dd];
    if (transposeOut) {
      bf16x8 pk;
#pragma unroll
      for (int r = 0; r < 8; r++) pk[r] = (bf16)(acc[t][r] + bv);
      int m0 = mt * 16 + 8 * half;
      *(bf16x8*)(dst + ((size_t)bh * 64 + dd) * 1024 + m0) = pk;
    } else {
#pragma unroll
      for (int r = 0; r < 8; r++) {
        int m = mt * 16 + r + 8 * half;
        dst[((size_t)bh * 1024 + m) * 64 + dd] = (bf16)(acc[t][r] + bv);
      }
    }
  }
}

// ---------------------------------------------------------------------------
// Projected spatial attention: per (b,h, 16-row q tile):
//   S = q @ k_s^T (16x1024), softmax rows, O = P @ v_s (16x64, split-K/wave)
// LDS: 64KB union { S f32[16][1024] ; red f32[8][16][64] }.
// ---------------------------------------------------------------------------
__global__ __launch_bounds__(256) void spat_attn_kernel(
    const bf16* __restrict__ q,    // [BH,1024,64]
    const bf16* __restrict__ ks,   // [BH,1024,64]
    const bf16* __restrict__ vsT,  // [BH,64,1024]
    const float* __restrict__ temp2,
    bf16* __restrict__ xs) {       // [B,1024,1024] : [b][n][h*64+dd]
  __shared__ union SM { float S[16][1024]; float red[8][16][64]; } sm;
  int bh = blockIdx.x >> 6;
  int qt = blockIdx.x & 63;
  int b = bh >> 4, h = bh & 15;
  float t2 = temp2[h];
  int wave = threadIdx.x >> 5; int l = lane_id(); int half = l >> 4; int cl = l & 15;
  const bf16* qb = q + ((size_t)bh * 1024 + qt * 16) * 64;
  const bf16* kb = ks + (size_t)bh * 1024 * 64;
  // phase 1: scores S[16][1024] (64 tiles, 8 per wave; K=64 fully unrolled)
  bf16x16 aq0 = load_a(qb, 64);
  bf16x16 aq1 = load_a(qb + 32, 64);
  for (int tn = wave; tn < 64; tn += 8) {
    const bf16* brow = kb + (size_t)(tn * 16) * 64;
    f32x8 acc = {};
    acc = wmma_bf16(aq0, load_b(brow, 64), acc);
    acc = wmma_bf16(aq1, load_b(brow + 32, 64), acc);
#pragma unroll
    for (int r = 0; r < 8; r++) sm.S[r + 8 * half][tn * 16 + cl] = acc[r] * t2;
  }
  __syncthreads();
  // phase 2: softmax; wave w: rows 2w(half=0), 2w+1(half=1); 16 lanes/row.
  {
    int row = 2 * wave + half;
    float* Sr = sm.S[row];
    int c0 = cl * 64;
    float m = -INFINITY;
    for (int j = 0; j < 64; j++) m = fmaxf(m, Sr[c0 + j]);
#pragma unroll
    for (int off = 1; off < 16; off <<= 1) m = fmaxf(m, __shfl_xor(m, off, 16));
    float s = 0.f;
    for (int j = 0; j < 64; j++) { float e = __expf(Sr[c0 + j] - m); Sr[c0 + j] = e; s += e; }
#pragma unroll
    for (int off = 1; off < 16; off <<= 1) s += __shfl_xor(s, off, 16);
    float inv = 1.f / s;
    for (int j = 0; j < 64; j++) Sr[c0 + j] *= inv;
  }
  __syncthreads();
  // phase 3: O = P @ v_s, split-K: this wave covers K in [wave*128, +128)
  const bf16* vb = vsT + (size_t)bh * 64 * 1024;
  f32x8 acc[4] = {};
  int k0 = wave * 128;
  for (int kk = k0; kk < k0 + 128; kk += 32) {
    bf16x16 a = load_a_f32(&sm.S[0][0] + kk, 1024);
#pragma unroll
    for (int t = 0; t < 4; t++) {
      bf16x16 bb = load_b(vb + (size_t)(t * 16) * 1024 + kk, 1024);
      acc[t] = wmma_bf16(a, bb, acc[t]);
    }
  }
  __syncthreads();   // all waves done reading S before we overwrite with red
#pragma unroll
  for (int t = 0; t < 4; t++)
#pragma unroll
    for (int r = 0; r < 8; r++)
      sm.red[wave][r + 8 * half][t * 16 + cl] = acc[t][r];
  __syncthreads();
  for (int e = threadIdx.x; e < 16 * 64; e += 256) {
    int m = e >> 6, dd = e & 63;
    float v = 0.f;
#pragma unroll
    for (int w2 = 0; w2 < 8; w2++) v += sm.red[w2][m][dd];
    int n = qt * 16 + m;
    xs[((size_t)b * 1024 + n) * 1024 + h * 64 + dd] = (bf16)v;
  }
}

// ---------------------------------------------------------------------------
// Output projection + exact-erf GELU: out[:, col_off:col_off+512] =
//   gelu(A @ W^T + bias).  Per-wave 32x64 tile, two-stage ping-pong.
// ---------------------------------------------------------------------------
__global__ __launch_bounds__(256) void out_proj_kernel(
    const bf16* __restrict__ A,     // [8192,1024]
    const bf16* __restrict__ W,     // [512,1024]
    const float* __restrict__ bias, // [512]
    float* __restrict__ out,        // [8192,1024]
    int col_off) {
  const int K = 1024;
  int wave = threadIdx.x >> 5;
  int job = blockIdx.x * 8 + wave;  // 2048 = 256 M-tiles * 8 strips
  int mt = job >> 3; int n0 = (job & 7) * 64;
  const bf16* A0 = A + (size_t)(mt * 32) * K;
  const bf16* A1 = A0 + (size_t)16 * K;
  const bf16* Bp = W + (size_t)n0 * K;
  f32x8 acc[8] = {};
  bf16x16 aE0, aE1, bE[4], aO0, aO1, bO[4];
  aE0 = load_a(A0, K);
  aE1 = load_a(A1, K);
#pragma unroll
  for (int t = 0; t < 4; t++) bE[t] = load_b(Bp + (size_t)(t * 16) * K, K);
  aO0 = load_a(A0 + 32, K);
  aO1 = load_a(A1 + 32, K);
#pragma unroll
  for (int t = 0; t < 4; t++) bO[t] = load_b(Bp + (size_t)(t * 16) * K + 32, K);
  for (int kk = 0; kk < K - 64; kk += 64) {
    __builtin_prefetch((const void*)(A0 + kk + 128), 0, 1);
    __builtin_prefetch((const void*)(A1 + kk + 128), 0, 1);
#pragma unroll
    for (int t = 0; t < 4; t++) acc[t]     = wmma_bf16(aE0, bE[t], acc[t]);
#pragma unroll
    for (int t = 0; t < 4; t++) acc[4 + t] = wmma_bf16(aE1, bE[t], acc[4 + t]);
    aE0 = load_a(A0 + kk + 64, K);
    aE1 = load_a(A1 + kk + 64, K);
#pragma unroll
    for (int t = 0; t < 4; t++) bE[t] = load_b(Bp + (size_t)(t * 16) * K + kk + 64, K);
#pragma unroll
    for (int t = 0; t < 4; t++) acc[t]     = wmma_bf16(aO0, bO[t], acc[t]);
#pragma unroll
    for (int t = 0; t < 4; t++) acc[4 + t] = wmma_bf16(aO1, bO[t], acc[4 + t]);
    aO0 = load_a(A0 + kk + 96, K);
    aO1 = load_a(A1 + kk + 96, K);
#pragma unroll
    for (int t = 0; t < 4; t++) bO[t] = load_b(Bp + (size_t)(t * 16) * K + kk + 96, K);
  }
#pragma unroll
  for (int t = 0; t < 4; t++) acc[t]     = wmma_bf16(aE0, bE[t], acc[t]);
#pragma unroll
  for (int t = 0; t < 4; t++) acc[4 + t] = wmma_bf16(aE1, bE[t], acc[4 + t]);
#pragma unroll
  for (int t = 0; t < 4; t++) acc[t]     = wmma_bf16(aO0, bO[t], acc[t]);
#pragma unroll
  for (int t = 0; t < 4; t++) acc[4 + t] = wmma_bf16(aO1, bO[t], acc[4 + t]);

  int l = lane_id(); int half = l >> 4; int cl = l & 15;
#pragma unroll
  for (int sub = 0; sub < 2; sub++) {
#pragma unroll
    for (int t = 0; t < 4; t++) {
      int c = n0 + t * 16 + cl;
      float bv = bias[c];
#pragma unroll
      for (int r = 0; r < 8; r++) {
        int m = mt * 32 + sub * 16 + r + 8 * half;
        float x = acc[sub * 4 + t][r] + bv;
        float g = 0.5f * x * (1.f + erff(x * 0.70710678118654752f));
        out[(size_t)m * 1024 + col_off + c] = g;
      }
    }
  }
}

// ---------------------------------------------------------------------------
extern "C" void kernel_launch(void* const* d_in, const int* in_sizes, int n_in,
                              void* d_out, int out_size, void* d_ws, size_t ws_size,
                              hipStream_t stream) {
  (void)in_sizes; (void)n_in; (void)out_size; (void)ws_size;
  const float* x      = (const float*)d_in[0];
  const float* w_qkvv = (const float*)d_in[1];
  const float* w_E    = (const float*)d_in[2];
  const float* b_E    = (const float*)d_in[3];
  const float* w_F    = (const float*)d_in[4];
  const float* b_F    = (const float*)d_in[5];
  const float* w_o1   = (const float*)d_in[6];
  const float* b_o1   = (const float*)d_in[7];
  const float* w_o2   = (const float*)d_in[8];
  const float* b_o2   = (const float*)d_in[9];
  const float* temp1  = (const float*)d_in[10];
  const float* temp2  = (const float*)d_in[11];
  float* out = (float*)d_out;

  const size_t BIG = 8388608;   // B*H*N*d == B*N*C elements
  char* p = (char*)d_ws;
  auto take = [&](size_t elems) {
    bf16* r = (bf16*)p;
    p += (elems * sizeof(bf16) + 255) & ~(size_t)255;
    return r;
  };
  bf16* xb   = take(BIG);
  bf16* wqb  = take(4194304);
  bf16* wEb  = take(4096);
  bf16* wFb  = take(4096);
  bf16* wo1b = take(524288);
  bf16* wo2b = take(524288);
  bf16* q    = take(BIG);
  bf16* qT   = take(BIG);
  bf16* k    = take(BIG);
  bf16* kT   = take(BIG);
  bf16* vca  = take(BIG);
  bf16* vsa  = take(BIG);
  bf16* ksP  = take(BIG);   // k_s
  bf16* vsT  = take(BIG);   // v_s transposed [BH,64,1024]
  bf16* xc   = take(BIG);
  bf16* xs   = take(BIG);

  auto cvt = [&](const float* src, bf16* dst, int n) {
    int blocks = (n + 255) / 256; if (blocks > 4096) blocks = 4096;
    cvt_f32_bf16<<<blocks, 256, 0, stream>>>(src, dst, n);
  };
  cvt(x,      xb,   (int)BIG);
  cvt(w_qkvv, wqb,  4194304);
  cvt(w_E,    wEb,  4096);
  cvt(w_F,    wFb,  4096);
  cvt(w_o1,   wo1b, 524288);
  cvt(w_o2,   wo2b, 524288);

  qkvv_kernel<<<2048, 256, 0, stream>>>(xb, wqb, q, qT, k, kT, vca, vsa);
  proj_ef_kernel<<<1024, 256, 0, stream>>>(k,   wEb, b_E, ksP, 0);
  proj_ef_kernel<<<1024, 256, 0, stream>>>(vsa, wFb, b_F, vsT, 1);
  chan_attn_kernel<<<128, 256, 0, stream>>>(qT, kT, vca, temp1, xc);
  spat_attn_kernel<<<8192, 256, 0, stream>>>(q, ksP, vsT, temp2, xs);
  out_proj_kernel<<<256, 256, 0, stream>>>(xs, wo1b, b_o1, out, 0);
  out_proj_kernel<<<256, 256, 0, stream>>>(xc, wo2b, b_o2, out, 512);
}
